// GTCNN_block_56813827391868
// MI455X (gfx1250) — matrix operand order repaired and checked by reading, not compiled, not run on hardware
//
#include <hip/hip_runtime.h>
#include <hip/hip_bf16.h>
#include <math.h>

#define B_    8
#define T_    8
#define S_    500
#define FIN_  32
#define FOUT_ 64
#define SP_   512                 // S padded to multiple of 32 for WMMA K-loop
#define XPAD_ 72                  // padded LDS row stride (elems): conflict-free ds_load_b128
#define ROWS_ (B_*T_*S_)          // 32000
#define BT_   (B_*T_)             // 64

typedef __attribute__((ext_vector_type(16))) __bf16 v16bf;
typedef __attribute__((ext_vector_type(8)))  __bf16 v8bf;
typedef __attribute__((ext_vector_type(8)))  float  v8f;

static __device__ inline v16bf cat2(v8bf lo, v8bf hi) {
  v16bf r;
#pragma unroll
  for (int i = 0; i < 8; ++i) { r[i] = lo[i]; r[i + 8] = hi[i]; }
  return r;
}

static __device__ inline v8bf zero_v8() {
  v8bf r;
#pragma unroll
  for (int i = 0; i < 8; ++i) r[i] = (__bf16)0.0f;
  return r;
}

static __device__ inline v8f wmma_bf(v16bf a, v16bf b, v8f c) {
  // D(f32 16x16) = A(bf16 16x32) * B(bf16 32x16) + C
  return __builtin_amdgcn_wmma_f32_16x16x32_bf16(
      /*neg_a=*/false, a, /*neg_b=*/false, b,
      /*c_mod=*/(short)0, c, /*reuse_a=*/false, /*reuse_b=*/false);
}

static __device__ inline void wait_asynccnt0() {
#if __has_builtin(__builtin_amdgcn_s_wait_asynccnt)
  __builtin_amdgcn_s_wait_asynccnt(0);
#else
  asm volatile("s_wait_asynccnt 0x0" ::: "memory");
#endif
}

// ---------------------------------------------------------------------------
// Prep: Adj_s -> bf16 zero-padded [512x512]; W1^T, H, W2^T -> bf16
// ---------------------------------------------------------------------------
__global__ void prep_kernel(const float* __restrict__ adjs,
                            const float* __restrict__ w1,
                            const float* __restrict__ h,
                            const float* __restrict__ w2,
                            __bf16* __restrict__ adjs_bf,
                            __bf16* __restrict__ w1t_bf,
                            __bf16* __restrict__ h_bf,
                            __bf16* __restrict__ w2t_bf) {
  int idx = blockIdx.x * blockDim.x + threadIdx.x;
  const int NADJ = SP_ * SP_;
  const int NW1  = FIN_ * FOUT_;
  const int NH   = 9 * FOUT_ * FOUT_;
  const int NW2  = FOUT_ * FOUT_;
  if (idx < NADJ) {
    int r = idx >> 9, c = idx & (SP_ - 1);
    float v = (r < S_ && c < S_) ? adjs[r * S_ + c] : 0.0f;
    adjs_bf[idx] = (__bf16)v;
  } else if (idx < NADJ + NW1) {
    int j = idx - NADJ; int k = j >> 6, n = j & 63;   // [K=fin][N=fout]
    w1t_bf[j] = (__bf16)w1[n * FIN_ + k];
  } else if (idx < NADJ + NW1 + NH) {
    int j = idx - NADJ - NW1;                          // H already [K][N] row-major
    h_bf[j] = (__bf16)h[j];
  } else if (idx < NADJ + NW1 + NH + NW2) {
    int j = idx - NADJ - NW1 - NH; int k = j >> 6, n = j & 63;
    w2t_bf[j] = (__bf16)w2[n * FOUT_ + k];
  }
}

// ---------------------------------------------------------------------------
// xf = x @ W1^T + b1   ([32000,32]@[32,64]); one wave = 16x64 C strip, K=32
// ---------------------------------------------------------------------------
__global__ void gemm_x_w1(const float* __restrict__ x,
                          const __bf16* __restrict__ w1t,
                          const float* __restrict__ b1,
                          __bf16* __restrict__ out) {
  int lane  = threadIdx.x & 31;
  int mtile = blockIdx.x * (blockDim.x >> 5) + (threadIdx.x >> 5); // 0..1999
  int m = lane & 15, half = lane >> 4;
  const float* xr = x + (size_t)(mtile * 16 + m) * FIN_;
  v16bf a;
#pragma unroll
  for (int i = 0; i < 8; ++i) {
    a[i]     = (__bf16)xr[half * 8 + i];
    a[i + 8] = (__bf16)xr[16 + half * 8 + i];
  }
  v8f acc[4] = {};
#pragma unroll
  for (int nt = 0; nt < 4; ++nt) {
    v16bf b;
    for (int i = 0; i < 8; ++i) {
      b[i]     = w1t[(half * 8 + i) * FOUT_ + nt * 16 + m];
      b[i + 8] = w1t[(16 + half * 8 + i) * FOUT_ + nt * 16 + m];
    }
    acc[nt] = wmma_bf(a, b, acc[nt]);
  }
  int ocol = lane & 15;
  size_t orow0 = (size_t)mtile * 16 + 8 * half;
#pragma unroll
  for (int nt = 0; nt < 4; ++nt)
#pragma unroll
    for (int r = 0; r < 8; ++r) {
      float v = acc[nt][r] + b1[nt * 16 + ocol];
      out[(orow0 + r) * FOUT_ + nt * 16 + ocol] = (__bf16)v;
    }
}

// ---------------------------------------------------------------------------
// P[b,t] = Adj_s @ X[b,t]  ([500,500]@[500,64])
// Block = 8 waves sharing one X[b,t] tile staged in LDS via
// global_load_async_to_lds_b128 (ASYNCcnt). Grid: 64 bt x 4 mgroups.
// ---------------------------------------------------------------------------
__global__ __launch_bounds__(256) void spatial_gemm(
    const __bf16* __restrict__ xin,
    const __bf16* __restrict__ adjs_bf,
    __bf16* __restrict__ pout) {
  __shared__ __bf16 Xs[SP_ * XPAD_];   // 512 rows x 72 elems = 72 KB

  int tid    = threadIdx.x;
  int bt     = blockIdx.x >> 2;        // 0..63
  int mgroup = blockIdx.x & 3;         // 0..3
  const __bf16* Xbt = xin + (size_t)bt * S_ * FOUT_;

  // Cooperative async fill: 512 rows x 8 chunks of 16B each = 4096 chunks.
  // The sole __shared__ array sits at LDS offset 0, so the LDS byte address
  // for chunk (row, c8) is (row*XPAD_ + c8*8)*2.
#pragma unroll
  for (int i = 0; i < 16; ++i) {
    int ch  = tid + i * 256;           // 0..4095
    int row = ch >> 3;
    int c8  = ch & 7;
    unsigned lds_off = (unsigned)(row * XPAD_ + c8 * 8) * 2u;
    if (row < S_) {
      unsigned long long ga =
          (unsigned long long)(size_t)(Xbt + (size_t)row * FOUT_ + c8 * 8);
      asm volatile("global_load_async_to_lds_b128 %0, %1, off"
                   :: "v"(lds_off), "v"(ga) : "memory");
    } else {
      *(v8bf*)(Xs + (size_t)row * XPAD_ + c8 * 8) = zero_v8();  // pad rows = 0
    }
  }
  wait_asynccnt0();      // each wave drains its own ASYNCcnt ...
  __syncthreads();       // ... then the block barrier makes LDS tile visible

  int lane   = tid & 31;
  int waveid = tid >> 5;
  int mtile  = mgroup * 8 + waveid;    // 0..31
  int m = lane & 15, half = lane >> 4;
  const __bf16* Ar = adjs_bf + (size_t)(mtile * 16 + m) * SP_;

  v8f acc[4] = {};
  for (int kc = 0; kc < SP_ / 32; ++kc) {
    int kbase = kc * 32;
    v16bf a = cat2(*(const v8bf*)(Ar + kbase + half * 8),
                   *(const v8bf*)(Ar + kbase + 16 + half * 8));
    const __bf16* Xr = Xs + (size_t)kbase * XPAD_;  // B k-chunk base (column n=lane%16 gathered)
#pragma unroll
    for (int nt = 0; nt < 4; ++nt) {
      v16bf b;
      for (int i = 0; i < 8; ++i) {
        b[i]     = Xr[(half * 8 + i) * XPAD_ + nt * 16 + m];
        b[i + 8] = Xr[(16 + half * 8 + i) * XPAD_ + nt * 16 + m];
      }
      acc[nt] = wmma_bf(a, b, acc[nt]);
    }
  }

  int ocol = lane & 15;
  int orow0 = mtile * 16 + 8 * half;
#pragma unroll
  for (int r = 0; r < 8; ++r) {
    int srow = orow0 + r;
    if (srow < S_) {
#pragma unroll
      for (int nt = 0; nt < 4; ++nt)
        pout[((size_t)bt * S_ + srow) * FOUT_ + nt * 16 + ocol] = (__bf16)acc[nt][r];
    }
  }
}

// ---------------------------------------------------------------------------
// Z = s0*X + s1*P + s2*(Adj_t over t)X + s3*(Adj_t over t)P
// ---------------------------------------------------------------------------
__global__ void temporal_combine(const __bf16* __restrict__ X,
                                 const __bf16* __restrict__ P,
                                 const float* __restrict__ adjt,
                                 const float* __restrict__ scoef,
                                 __bf16* __restrict__ Z) {
  int idx = blockIdx.x * blockDim.x + threadIdx.x;
  if (idx >= B_ * T_ * S_ * FOUT_) return;
  int f = idx & 63;
  int rest = idx >> 6;
  int s  = rest % S_;
  int bt = rest / S_;
  int t = bt & 7, b = bt >> 3;
  float c0 = scoef[0], c1 = scoef[1], c2 = scoef[2], c3 = scoef[3];
  size_t base_b = (size_t)b * T_ * S_ * FOUT_;
  size_t sf = (size_t)s * FOUT_ + f;
  float xq = 0.f, pq = 0.f;
#pragma unroll
  for (int tp = 0; tp < T_; ++tp) {
    float wt = adjt[t * T_ + tp];
    size_t off = base_b + (size_t)tp * S_ * FOUT_ + sf;
    xq += wt * (float)X[off];
    pq += wt * (float)P[off];
  }
  size_t self = base_b + (size_t)t * S_ * FOUT_ + sf;
  float z = c0 * (float)X[self] + c1 * (float)P[self] + c2 * xq + c3 * pq;
  Z[self] = (__bf16)z;
}

// ---------------------------------------------------------------------------
// out = tanh(z0@H0 + z1@H1 + z2@H2)   (K=64 -> 2 chunks, 3 terms, 4 N-tiles)
// ---------------------------------------------------------------------------
__global__ void feature_gemm_tanh(const __bf16* __restrict__ z0,
                                  const __bf16* __restrict__ z1,
                                  const __bf16* __restrict__ z2,
                                  const __bf16* __restrict__ Hl,
                                  __bf16* __restrict__ out) {
  int lane  = threadIdx.x & 31;
  int mtile = blockIdx.x * (blockDim.x >> 5) + (threadIdx.x >> 5); // 0..1999
  int m = lane & 15, half = lane >> 4;
  size_t row = (size_t)mtile * 16 + m;
  const __bf16* zs[3] = { z0 + row * FOUT_, z1 + row * FOUT_, z2 + row * FOUT_ };
  v8f acc[4] = {};
#pragma unroll
  for (int j = 0; j < 3; ++j) {
    const __bf16* Hj = Hl + j * FOUT_ * FOUT_;
#pragma unroll
    for (int kc = 0; kc < 2; ++kc) {
      int kbase = kc * 32;
      v16bf a = cat2(*(const v8bf*)(zs[j] + kbase + half * 8),
                     *(const v8bf*)(zs[j] + kbase + 16 + half * 8));
#pragma unroll
      for (int nt = 0; nt < 4; ++nt) {
        v16bf b;
        for (int i = 0; i < 8; ++i) {
          b[i]     = Hj[(size_t)(kbase + half * 8 + i) * FOUT_ + nt * 16 + m];
          b[i + 8] = Hj[(size_t)(kbase + 16 + half * 8 + i) * FOUT_ + nt * 16 + m];
        }
        acc[nt] = wmma_bf(a, b, acc[nt]);
      }
    }
  }
  int ocol = lane & 15;
  size_t orow0 = (size_t)mtile * 16 + 8 * half;
#pragma unroll
  for (int nt = 0; nt < 4; ++nt)
#pragma unroll
    for (int r = 0; r < 8; ++r)
      out[(orow0 + r) * FOUT_ + nt * 16 + ocol] = (__bf16)tanhf(acc[nt][r]);
}

// ---------------------------------------------------------------------------
// xm[b,s,f] = sum_t merge[t]*xf[b,t,s,f]   (merge commutes with affine W2)
// ---------------------------------------------------------------------------
__global__ void merge_t(const __bf16* __restrict__ xf,
                        const float* __restrict__ mrg,
                        __bf16* __restrict__ xm) {
  int idx = blockIdx.x * blockDim.x + threadIdx.x;
  if (idx >= B_ * S_ * FOUT_) return;
  int f = idx & 63;
  int rest = idx >> 6;
  int s = rest % S_;
  int b = rest / S_;
  float acc = 0.f;
#pragma unroll
  for (int t = 0; t < T_; ++t)
    acc += mrg[t] * (float)xf[(((size_t)b * T_ + t) * S_ + s) * FOUT_ + f];
  xm[idx] = (__bf16)acc;
}

// ---------------------------------------------------------------------------
// out = xm @ W2^T + (sum_t merge[t]) * b2   -> f32 straight into d_out
// ---------------------------------------------------------------------------
__global__ void final_gemm(const __bf16* __restrict__ xm,
                           const __bf16* __restrict__ w2t,
                           const float* __restrict__ b2,
                           const float* __restrict__ mrg,
                           float* __restrict__ out) {
  int lane  = threadIdx.x & 31;
  int mtile = blockIdx.x * (blockDim.x >> 5) + (threadIdx.x >> 5);
  if (mtile >= (B_ * S_) / 16) return;   // 250 tiles
  int m = lane & 15, half = lane >> 4;
  const __bf16* xr = xm + (size_t)(mtile * 16 + m) * FOUT_;
  float msum = 0.f;
#pragma unroll
  for (int t = 0; t < T_; ++t) msum += mrg[t];
  v8f acc[4] = {};
#pragma unroll
  for (int kc = 0; kc < 2; ++kc) {
    int kbase = kc * 32;
    v16bf a = cat2(*(const v8bf*)(xr + kbase + half * 8),
                   *(const v8bf*)(xr + kbase + 16 + half * 8));
#pragma unroll
    for (int nt = 0; nt < 4; ++nt) {
      v16bf b;
      for (int i = 0; i < 8; ++i) {
        b[i]     = w2t[(size_t)(kbase + half * 8 + i) * FOUT_ + nt * 16 + m];
        b[i + 8] = w2t[(size_t)(kbase + 16 + half * 8 + i) * FOUT_ + nt * 16 + m];
      }
      acc[nt] = wmma_bf(a, b, acc[nt]);
    }
  }
  int ocol = lane & 15;
  size_t orow0 = (size_t)mtile * 16 + 8 * half;
#pragma unroll
  for (int nt = 0; nt < 4; ++nt)
#pragma unroll
    for (int r = 0; r < 8; ++r)
      out[(orow0 + r) * FOUT_ + nt * 16 + ocol] = acc[nt][r] + msum * b2[nt * 16 + ocol];
}

// ---------------------------------------------------------------------------
extern "C" void kernel_launch(void* const* d_in, const int* in_sizes, int n_in,
                              void* d_out, int out_size, void* d_ws, size_t ws_size,
                              hipStream_t stream) {
  (void)in_sizes; (void)n_in; (void)out_size; (void)ws_size;
  const float* x     = (const float*)d_in[0];
  const float* adjt  = (const float*)d_in[1];
  const float* adjs  = (const float*)d_in[2];
  const float* scoef = (const float*)d_in[3];
  const float* H     = (const float*)d_in[4];
  const float* W1    = (const float*)d_in[5];
  const float* b1    = (const float*)d_in[6];
  const float* W2    = (const float*)d_in[7];
  const float* b2    = (const float*)d_in[8];
  const float* mrg   = (const float*)d_in[9];
  float* out = (float*)d_out;

  char* ws = (char*)d_ws;
  size_t off = 0;
  auto carve = [&](size_t bytes) -> char* {
    char* p = ws + off;
    off += (bytes + 255) & ~(size_t)255;
    return p;
  };
  __bf16* adjs_bf = (__bf16*)carve((size_t)SP_ * SP_ * 2);
  __bf16* w1t_bf  = (__bf16*)carve((size_t)FIN_ * FOUT_ * 2);
  __bf16* h_bf    = (__bf16*)carve((size_t)9 * FOUT_ * FOUT_ * 2);
  __bf16* w2t_bf  = (__bf16*)carve((size_t)FOUT_ * FOUT_ * 2);
  __bf16* act[4];
  for (int i = 0; i < 4; ++i) act[i] = (__bf16*)carve((size_t)ROWS_ * FOUT_ * 2);
  __bf16* Pbuf = (__bf16*)carve((size_t)ROWS_ * FOUT_ * 2);
  __bf16* xm   = (__bf16*)carve((size_t)B_ * S_ * FOUT_ * 2);

  prep_kernel<<<1192, 256, 0, stream>>>(adjs, W1, H, W2, adjs_bf, w1t_bf, h_bf, w2t_bf);
  gemm_x_w1<<<250, 256, 0, stream>>>(x, w1t_bf, b1, act[0]);

  int cur = 0;
  for (int l = 0; l < 3; ++l) {
    int z1i = (cur + 1) & 3, z2i = (cur + 2) & 3, oi = (cur + 3) & 3;
    spatial_gemm<<<256, 256, 0, stream>>>(act[cur], adjs_bf, Pbuf);
    temporal_combine<<<8000, 256, 0, stream>>>(act[cur], Pbuf, adjt, scoef, act[z1i]);
    spatial_gemm<<<256, 256, 0, stream>>>(act[z1i], adjs_bf, Pbuf);
    temporal_combine<<<8000, 256, 0, stream>>>(act[z1i], Pbuf, adjt, scoef, act[z2i]);
    feature_gemm_tanh<<<250, 256, 0, stream>>>(act[cur], act[z1i], act[z2i],
                                               h_bf + (size_t)l * 3 * FOUT_ * FOUT_,
                                               act[oi]);
    cur = oi;
  }
  merge_t<<<1000, 256, 0, stream>>>(act[cur], mrg, xm);
  final_gemm<<<32, 256, 0, stream>>>(xm, w2t_bf, b2, mrg, out);
}
